// LocationSensitiveAttention_67422396613102
// MI455X (gfx1250) — matrix operand order repaired and checked by reading, not compiled, not run on hardware
//
#include <hip/hip_runtime.h>
#include <hip/hip_bf16.h>
#include <math.h>

// ---------------------------------------------------------------------------
// LocationSensitiveAttention for MI455X (gfx1250, wave32, WMMA + async-LDS)
//
// Shapes: R=1024 E=512 A=128 F=32 Kc=31 B=64 T=2048
// Outputs (concat): attention_output (B,E) f32  then attention_weights (B,T)
// ---------------------------------------------------------------------------

#define R_DIM 1024
#define E_DIM 512
#define A_DIM 128
#define F_DIM 32
#define KC    31
#define B_DIM 64
#define T_DIM 2048

typedef __attribute__((ext_vector_type(16))) __bf16 v16bf;
typedef __attribute__((ext_vector_type(8)))  float  v8f;

// ---------------------------------------------------------------------------
// K0: one-shot weight conversion f32 -> bf16, transposed to [N][K] so the
// main kernel's B-fragment staging is a raw byte copy (async-LDS friendly).
// ---------------------------------------------------------------------------
__global__ void lsa_wcvt_kernel(const float* __restrict__ Wmem,
                                const float* __restrict__ Wloc,
                                __bf16* __restrict__ WbfT,
                                __bf16* __restrict__ WlocT) {
  const int stride = gridDim.x * blockDim.x;
  int idx = blockIdx.x * blockDim.x + threadIdx.x;
  for (int i = idx; i < E_DIM * A_DIM; i += stride) {
    const int n = i & (A_DIM - 1), k = i >> 7;
    WbfT[(size_t)n * E_DIM + k] = (__bf16)Wmem[(size_t)k * A_DIM + n];
  }
  for (int i = idx; i < F_DIM * A_DIM; i += stride) {
    const int f = i & (F_DIM - 1), n = i >> 5;
    WlocT[(size_t)n * F_DIM + f] = (__bf16)Wloc[(size_t)f * A_DIM + n];
  }
}

// ---------------------------------------------------------------------------
// K1: pq[b,a] = attention_hidden[b,:] @ W_query[:,a]
// ---------------------------------------------------------------------------
__global__ void lsa_pq_kernel(const float* __restrict__ hidden,
                              const float* __restrict__ Wq,
                              float* __restrict__ pq) {
  __shared__ float h_s[R_DIM];
  const int b = blockIdx.x;
  const int a = threadIdx.x;
  for (int i = threadIdx.x; i < R_DIM; i += blockDim.x)
    h_s[i] = hidden[(size_t)b * R_DIM + i];
  __syncthreads();
  float s = 0.f;
#pragma unroll 8
  for (int r = 0; r < R_DIM; ++r)
    s += h_s[r] * Wq[(size_t)r * A_DIM + a];
  pq[(size_t)b * A_DIM + a] = s;
}

// ---------------------------------------------------------------------------
// K2: loc[b,t,f] = conv1d(attention_weights_cat), same padding, bf16 (b,t,f)
// ---------------------------------------------------------------------------
__global__ void lsa_conv_kernel(const float* __restrict__ awc,   // (B,2,T)
                                const float* __restrict__ convw, // (F,2,KC)
                                __bf16* __restrict__ loc) {      // (B,T,F)
  __shared__ float xs[2 * (128 + KC - 1)];
  __shared__ float ws[F_DIM * 2 * KC];
  const int b  = blockIdx.y;
  const int t0 = blockIdx.x * 128;
  const int tid = threadIdx.x;
  const int W = 128 + KC - 1;

  for (int idx = tid; idx < 2 * W; idx += 128) {
    const int c = idx / W, pos = idx % W;
    const int src = t0 - (KC - 1) / 2 + pos;
    xs[c * W + pos] = (src >= 0 && src < T_DIM)
                        ? awc[((size_t)b * 2 + c) * T_DIM + src] : 0.f;
  }
  for (int idx = tid; idx < F_DIM * 2 * KC; idx += 128)
    ws[idx] = convw[idx];
  __syncthreads();

  const size_t row = ((size_t)b * T_DIM + t0 + tid) * F_DIM;
#pragma unroll 4
  for (int f = 0; f < F_DIM; ++f) {
    float s = 0.f;
#pragma unroll
    for (int c = 0; c < 2; ++c)
#pragma unroll
      for (int k = 0; k < KC; ++k)
        s += xs[c * W + tid + k] * ws[f * 2 * KC + c * KC + k];
    loc[row + f] = (__bf16)s;
  }
}

// ---------------------------------------------------------------------------
// helpers: async copy of 16B global -> LDS (gfx1250, tracked by ASYNCcnt)
// ---------------------------------------------------------------------------
__device__ __forceinline__ void async_copy16(const __bf16* g, __bf16* lds_dst) {
  const unsigned loff = (unsigned)(size_t)lds_dst;   // low 32 bits = LDS offset
  asm volatile("global_load_async_to_lds_b128 %0, %1, off"
               :: "v"(loff), "v"(g) : "memory");
}
__device__ __forceinline__ void async_wait0() {
  asm volatile("s_wait_asynccnt 0" ::: "memory");
}

// ---------------------------------------------------------------------------
// K3: WMMA core.  Each wave owns a 32(T) x 128(A) tile: acc[2][8] f32.
// __launch_bounds__(256,1): 8 waves/block, 2 waves/SIMD -> up to ~512 VGPRs
// per wave so the 128 accumulator VGPRs + fragments live entirely in regs
// (round-2 build spilled accs to scratch inside the hot loop).
// B fragments double-buffered: load j+1 while two WMMAs consume j.
// ---------------------------------------------------------------------------
__global__ void __launch_bounds__(256, 1)
lsa_wmma_kernel(const float*  __restrict__ memory, // (B,T,E)
                const __bf16* __restrict__ WbfT,   // [128][512]
                const __bf16* __restrict__ WlocT,  // [128][32]
                const float*  __restrict__ pq,     // (B,A)
                const __bf16* __restrict__ loc,    // (B,T,F)
                const float*  __restrict__ v_w,    // (A)
                const float*  __restrict__ v_b,    // (1)
                const int*    __restrict__ maskp,  // (B,T)
                float* __restrict__ align_out) {   // (B,T)
  __shared__ __bf16 Wlds[A_DIM * 128];     // 32KB: chunk [n][kl]
  __shared__ __bf16 Wloc_s[A_DIM * F_DIM]; // 8KB:  [n][f]
  __shared__ float  pq_s[A_DIM];

  const int b    = blockIdx.y;
  const int tid  = threadIdx.x;
  const int wave = tid >> 5;
  const int lane = tid & 31;
  const int nl   = lane & 15;
  const int hi   = lane >> 4;

  // stage W_loc tile asynchronously (raw bf16 copy; 512 x 16B segments)
#pragma unroll
  for (int i = 0; i < 2; ++i) {
    const int s = tid + i * 256;
    async_copy16(WlocT + s * 8, &Wloc_s[s * 8]);
  }
  if (tid < A_DIM) pq_s[tid] = pq[(size_t)b * A_DIM + tid];

  const int t0 = blockIdx.x * 256 + wave * 32;   // wave's first T row
  const float* rowA0 = memory + ((size_t)b * T_DIM + t0 + nl) * E_DIM;
  const float* rowA1 = rowA0 + (size_t)16 * E_DIM;

  v8f acc[2][8];
#pragma unroll
  for (int m = 0; m < 2; ++m)
#pragma unroll
    for (int j = 0; j < 8; ++j) acc[m][j] = (v8f)0.f;

  // ---- K loop over E in 128-wide LDS chunks ----
  for (int kc = 0; kc < E_DIM; kc += 128) {
    __syncthreads();                // everyone done reading previous chunk
#pragma unroll
    for (int i = 0; i < 8; ++i) {  // async-stage chunk: 2048 x 16B segments
      const int s   = tid + i * 256;
      const int n   = s >> 4, sub = s & 15;
      async_copy16(WbfT + (size_t)n * E_DIM + kc + sub * 8,
                   &Wlds[n * 128 + sub * 8]);
    }
    async_wait0();                  // drain this wave's ASYNCcnt
    __syncthreads();                // chunk visible to all waves

    if (kc + 128 < E_DIM) {
      __builtin_prefetch(rowA0 + kc + 128, 0, 1);  // global_prefetch_b8
      __builtin_prefetch(rowA1 + kc + 128, 0, 1);
    }

#pragma unroll
    for (int kk = 0; kk < 128; kk += 32) {
      const int kbase = kc + kk;
      // Two A fragments (rows t0.. / t0+16..), 16-bit A striping:
      //   e<8 : K = kbase + hi*8 + e ; e>=8 : K = kbase + 16 + hi*8 + (e-8)
      v16bf a0, a1;
      {
        const float4 f0 = *(const float4*)(rowA0 + kbase + hi * 8);
        const float4 f1 = *(const float4*)(rowA0 + kbase + hi * 8 + 4);
        const float4 f2 = *(const float4*)(rowA0 + kbase + 16 + hi * 8);
        const float4 f3 = *(const float4*)(rowA0 + kbase + 16 + hi * 8 + 4);
        a0[0]=(__bf16)f0.x; a0[1]=(__bf16)f0.y; a0[2]=(__bf16)f0.z; a0[3]=(__bf16)f0.w;
        a0[4]=(__bf16)f1.x; a0[5]=(__bf16)f1.y; a0[6]=(__bf16)f1.z; a0[7]=(__bf16)f1.w;
        a0[8]=(__bf16)f2.x; a0[9]=(__bf16)f2.y; a0[10]=(__bf16)f2.z; a0[11]=(__bf16)f2.w;
        a0[12]=(__bf16)f3.x; a0[13]=(__bf16)f3.y; a0[14]=(__bf16)f3.z; a0[15]=(__bf16)f3.w;
      }
      {
        const float4 f0 = *(const float4*)(rowA1 + kbase + hi * 8);
        const float4 f1 = *(const float4*)(rowA1 + kbase + hi * 8 + 4);
        const float4 f2 = *(const float4*)(rowA1 + kbase + 16 + hi * 8);
        const float4 f3 = *(const float4*)(rowA1 + kbase + 16 + hi * 8 + 4);
        a1[0]=(__bf16)f0.x; a1[1]=(__bf16)f0.y; a1[2]=(__bf16)f0.z; a1[3]=(__bf16)f0.w;
        a1[4]=(__bf16)f1.x; a1[5]=(__bf16)f1.y; a1[6]=(__bf16)f1.z; a1[7]=(__bf16)f1.w;
        a1[8]=(__bf16)f2.x; a1[9]=(__bf16)f2.y; a1[10]=(__bf16)f2.z; a1[11]=(__bf16)f2.w;
        a1[12]=(__bf16)f3.x; a1[13]=(__bf16)f3.y; a1[14]=(__bf16)f3.z; a1[15]=(__bf16)f3.w;
      }
      // B fragments, double-buffered: N = j*16 + nl, K = kk + hi*8 + (e&7) + (e>>3)*16
      v16bf bmcur, bmnext;
      {
        const __bf16* wp = &Wlds[(0 * 16 + nl) * 128 + kk + hi * 8];
#pragma unroll
        for (int e = 0; e < 16; ++e) bmcur[e] = wp[(e & 7) + ((e >> 3) << 4)];
      }
#pragma unroll
      for (int j = 0; j < 8; ++j) {
        if (j < 7) {
          const __bf16* wp = &Wlds[((j + 1) * 16 + nl) * 128 + kk + hi * 8];
#pragma unroll
          for (int e = 0; e < 16; ++e) bmnext[e] = wp[(e & 7) + ((e >> 3) << 4)];
        }
        acc[0][j] = __builtin_amdgcn_wmma_f32_16x16x32_bf16(
            false, a0, false, bmcur, (short)0, acc[0][j], false, false);
        acc[1][j] = __builtin_amdgcn_wmma_f32_16x16x32_bf16(
            false, a1, false, bmcur, (short)0, acc[1][j], false, false);
        bmcur = bmnext;
      }
    }
  }

  // ---- location term: K=32 (=F), one WMMA pair per N tile ----
  {
    const __bf16* lp0 = loc + ((size_t)b * T_DIM + t0 + nl) * F_DIM;
    const __bf16* lp1 = lp0 + (size_t)16 * F_DIM;
    v16bf a0, a1;
#pragma unroll
    for (int e = 0; e < 8; ++e) {
      a0[e] = lp0[hi * 16 + e];      a0[8 + e] = lp0[hi * 16 + 8 + e];
      a1[e] = lp1[hi * 16 + e];      a1[8 + e] = lp1[hi * 16 + 8 + e];
    }
#pragma unroll
    for (int j = 0; j < 8; ++j) {
      const __bf16* wp = &Wloc_s[(j * 16 + nl) * F_DIM + hi * 16];
      v16bf bm;
#pragma unroll
      for (int e = 0; e < 16; ++e) bm[e] = wp[e];
      acc[0][j] = __builtin_amdgcn_wmma_f32_16x16x32_bf16(
          false, a0, false, bm, (short)0, acc[0][j], false, false);
      acc[1][j] = __builtin_amdgcn_wmma_f32_16x16x32_bf16(
          false, a1, false, bm, (short)0, acc[1][j], false, false);
    }
  }

  // ---- epilogue: energies = sum_n tanh(acc + pq[n]) * v_w[n] ----
  // value (M = m*16 + r + hi*8, N = j*16 + nl) lives in acc[m][j][r]
  float rowsum[2][8];
#pragma unroll
  for (int m = 0; m < 2; ++m)
#pragma unroll
    for (int r = 0; r < 8; ++r) rowsum[m][r] = 0.f;
#pragma unroll
  for (int j = 0; j < 8; ++j) {
    const float vw  = v_w[j * 16 + nl];
    const float pqv = pq_s[j * 16 + nl];
#pragma unroll
    for (int m = 0; m < 2; ++m)
#pragma unroll
      for (int r = 0; r < 8; ++r)
        rowsum[m][r] += tanhf(acc[m][j][r] + pqv) * vw;
  }
#pragma unroll
  for (int m = 0; m < 2; ++m)
#pragma unroll
    for (int r = 0; r < 8; ++r) {
#pragma unroll
      for (int x = 1; x < 16; x <<= 1)
        rowsum[m][r] += __shfl_xor(rowsum[m][r], x, 32);
    }
  if (nl == 0) {
    const float vb = v_b[0];
#pragma unroll
    for (int m = 0; m < 2; ++m)
#pragma unroll
      for (int r = 0; r < 8; ++r) {
        const int t = t0 + m * 16 + r + hi * 8;
        const float al = rowsum[m][r] + vb
                       + (float)maskp[(size_t)b * T_DIM + t] * -1e25f;
        align_out[(size_t)b * T_DIM + t] = al;
      }
  }
}

// ---------------------------------------------------------------------------
// K4: softmax -> attention_weights ; raw alignments x memory -> output
// ---------------------------------------------------------------------------
__global__ void lsa_out_kernel(const float* __restrict__ align,
                               const float* __restrict__ memory,
                               float* __restrict__ out_att,
                               float* __restrict__ out_w) {
  __shared__ float al_s[T_DIM];
  __shared__ float red[256];
  const int b = blockIdx.x;
  const int tid = threadIdx.x;

  float lmax = -3.0e38f;
  for (int t = tid; t < T_DIM; t += 256) {
    const float a = align[(size_t)b * T_DIM + t];
    al_s[t] = a;
    lmax = fmaxf(lmax, a);
  }
  red[tid] = lmax; __syncthreads();
  for (int s = 128; s > 0; s >>= 1) {
    if (tid < s) red[tid] = fmaxf(red[tid], red[tid + s]);
    __syncthreads();
  }
  const float rmax = red[0]; __syncthreads();

  float lsum = 0.f;
  for (int t = tid; t < T_DIM; t += 256) lsum += expf(al_s[t] - rmax);
  red[tid] = lsum; __syncthreads();
  for (int s = 128; s > 0; s >>= 1) {
    if (tid < s) red[tid] += red[tid + s];
    __syncthreads();
  }
  const float inv = 1.f / red[0]; __syncthreads();

  for (int t = tid; t < T_DIM; t += 256)
    out_w[(size_t)b * T_DIM + t] = expf(al_s[t] - rmax) * inv;

  float acc0 = 0.f, acc1 = 0.f;
  const float* mb = memory + (size_t)b * T_DIM * E_DIM;
#pragma unroll 4
  for (int t = 0; t < T_DIM; ++t) {
    const float a = al_s[t];
    acc0 += a * mb[(size_t)t * E_DIM + tid];
    acc1 += a * mb[(size_t)t * E_DIM + tid + 256];
  }
  out_att[(size_t)b * E_DIM + tid]       = acc0;
  out_att[(size_t)b * E_DIM + tid + 256] = acc1;
}

// ---------------------------------------------------------------------------
extern "C" void kernel_launch(void* const* d_in, const int* in_sizes, int n_in,
                              void* d_out, int out_size, void* d_ws, size_t ws_size,
                              hipStream_t stream) {
  const float* hidden = (const float*)d_in[0];
  const float* memory = (const float*)d_in[1];
  const float* awc    = (const float*)d_in[2];
  const int*   maskp  = (const int*)  d_in[3];
  const float* Wq     = (const float*)d_in[4];
  const float* Wmem   = (const float*)d_in[5];
  const float* convw  = (const float*)d_in[6];
  const float* Wloc   = (const float*)d_in[7];
  const float* v_w    = (const float*)d_in[8];
  const float* v_b    = (const float*)d_in[9];

  float* out_att = (float*)d_out;
  float* out_w   = (float*)d_out + (size_t)B_DIM * E_DIM;

  // ws: pq (32KB) | align (512KB) | loc bf16 (8MB) | WbfT (128KB) | WlocT (8KB)
  char* ws = (char*)d_ws;
  size_t off = 0;
  float*  pq    = (float*)(ws + off);  off += (size_t)B_DIM * A_DIM * 4;
  float*  align = (float*)(ws + off);  off += (size_t)B_DIM * T_DIM * 4;
  __bf16* loc   = (__bf16*)(ws + off); off += (size_t)B_DIM * T_DIM * F_DIM * 2;
  __bf16* WbfT  = (__bf16*)(ws + off); off += (size_t)E_DIM * A_DIM * 2;
  __bf16* WlocT = (__bf16*)(ws + off);

  lsa_wcvt_kernel<<<128, 256, 0, stream>>>(Wmem, Wloc, WbfT, WlocT);
  lsa_pq_kernel<<<B_DIM, 128, 0, stream>>>(hidden, Wq, pq);
  lsa_conv_kernel<<<dim3(T_DIM / 128, B_DIM), 128, 0, stream>>>(awc, convw, loc);
  lsa_wmma_kernel<<<dim3(T_DIM / 256, B_DIM), 256, 0, stream>>>(
      memory, WbfT, WlocT, pq, loc, v_w, v_b, maskp, align);
  lsa_out_kernel<<<B_DIM, 256, 0, stream>>>(align, memory, out_att, out_w);
}
